// TrajectronPooling_46634754900237
// MI455X (gfx1250) — matrix operand order repaired and checked by reading, not compiled, not run on hardware
//
#include <hip/hip_runtime.h>
#include <hip/hip_bf16.h>

typedef _Float16 v16h __attribute__((ext_vector_type(16)));
typedef float    v8f  __attribute__((ext_vector_type(8)));

#define N_PED 131072
#define HID   256
#define DOUT  32
#define GATES 1024   // 4*HID
#define MROWS 64     // rows per block in the fused kernel (4 M-tiles)

#define USE_ASYNC_LDS 1

// ---- fast transcendentals on CDNA5 TRANS units ----
#define LOG2E_F 1.44269504088896340736f

__device__ __forceinline__ float fast_exp(float x) {
    return __builtin_amdgcn_exp2f(x * LOG2E_F);     // v_exp_f32
}
__device__ __forceinline__ float fast_rcp(float x) {
    return __builtin_amdgcn_rcpf(x);                // v_rcp_f32
}
__device__ __forceinline__ float fast_sigmoid(float x) {
    return fast_rcp(1.0f + fast_exp(-x));
}
__device__ __forceinline__ float fast_tanh(float x) {
#if __has_builtin(__builtin_amdgcn_tanhf)
    return __builtin_amdgcn_tanhf(x);               // v_tanh_f32 (gfx1250)
#else
    // tanh(x) = 1 - 2/(e^{2x}+1)
    return 1.0f - 2.0f * fast_rcp(fast_exp(2.0f * x) + 1.0f);
#endif
}

// ---- Fragment assembly from two aligned 16-byte chunks ----
// Works for both global and LDS pointers (compiler picks global_load_b128 /
// ds_load_b128). Chunks must be 16B aligned.
__device__ __forceinline__ v16h load_frag2(const _Float16* p0, const _Float16* p1) {
    union { uint4 q[2]; v16h h; } u;
    u.q[0] = *(const uint4*)p0;
    u.q[1] = *(const uint4*)p1;
    return u.h;
}

// ---- A-fragment: 16-bit A matrix 16x32 (MxK), row-major source ----
// Lane l16 = row M. Elements 0..7  = K = kofs + half*8 + (0..7)
//            Elements 8..15 = K = kofs + 16 + half*8 + (0..7)
__device__ __forceinline__ v16h load_a(const _Float16* __restrict__ row,
                                       int half, int kofs) {
    return load_frag2(row + kofs + half * 8, row + kofs + 16 + half * 8);
}

// ---- B-fragment: 16-bit B matrix 32x16 (KxN); weight row-major W[n][k] ----
// Lane col n; Elements 0..7 = K = kofs + half*8 + (0..7), 8..15 = K = kofs + 16 + half*8 + (0..7).
__device__ __forceinline__ v16h load_b(const _Float16* __restrict__ w,
                                       int ldk, int n, int half, int kofs) {
    const _Float16* p = w + (long)n * ldk + kofs + half * 8;
    return load_frag2(p, p + 16);
}

// ---------------- Pass 1: per-block partial sums of states ----------------
__global__ __launch_bounds__(256) void traj_reduce1(
    const float* __restrict__ obs1, const float* __restrict__ obs2,
    float* __restrict__ partials, int n) {
    __shared__ float s[256][4];
    float a0 = 0.f, a1 = 0.f, a2 = 0.f, a3 = 0.f;
    for (int i = blockIdx.x * blockDim.x + threadIdx.x; i < n;
         i += gridDim.x * blockDim.x) {
        float o1x = obs1[2 * i], o1y = obs1[2 * i + 1];
        float o2x = obs2[2 * i], o2y = obs2[2 * i + 1];
        a0 += o2x; a1 += o2y; a2 += o2x - o1x; a3 += o2y - o1y;
    }
    int t = threadIdx.x;
    s[t][0] = a0; s[t][1] = a1; s[t][2] = a2; s[t][3] = a3;
    __syncthreads();
    for (int off = 128; off > 0; off >>= 1) {
        if (t < off) {
#pragma unroll
            for (int c = 0; c < 4; ++c) s[t][c] += s[t + off][c];
        }
        __syncthreads();
    }
    if (t == 0) {
#pragma unroll
        for (int c = 0; c < 4; ++c) partials[blockIdx.x * 4 + c] = s[0][c];
    }
}

// ---------------- Pass 2: deterministic final sum ----------------
__global__ void traj_reduce2(const float* __restrict__ partials,
                             float* __restrict__ totals) {
    int c = threadIdx.x;
    if (c < 4) {
        float a = 0.f;
        for (int b = 0; b < 256; ++b) a += partials[b * 4 + c];
        totals[c] = a;
    }
}

// ---------------- Pass 3: embedding MLP -> x (f16) ----------------
__global__ __launch_bounds__(256) void traj_embed(
    const float* __restrict__ obs1, const float* __restrict__ obs2,
    const float* __restrict__ totals,
    const float* __restrict__ W_emb, const float* __restrict__ b_emb,
    _Float16* __restrict__ xf, int n) {
    int i = blockIdx.x * blockDim.x + threadIdx.x;
    if (i >= n) return;
    float o1x = obs1[2 * i], o1y = obs1[2 * i + 1];
    float o2x = obs2[2 * i], o2y = obs2[2 * i + 1];
    float st[4] = {o2x, o2y, o2x - o1x, o2y - o1y};
    float g[8];
#pragma unroll
    for (int k = 0; k < 4; ++k) {
        g[k]     = st[k];
        g[k + 4] = totals[k] - st[k];
    }
#pragma unroll
    for (int d = 0; d < DOUT; ++d) {
        float acc = b_emb[d];
#pragma unroll
        for (int k = 0; k < 8; ++k) acc += g[k] * W_emb[d * 8 + k];
        xf[(long)i * DOUT + d] = (_Float16)fmaxf(acc, 0.f);
    }
}

// ---------------- Pass 4: generic f32 -> f16 conversion ----------------
__global__ __launch_bounds__(256) void traj_cvt_f16(
    const float* __restrict__ in, _Float16* __restrict__ out, long n) {
    long stride = (long)gridDim.x * blockDim.x;
    for (long i = (long)blockIdx.x * blockDim.x + threadIdx.x; i < n; i += stride)
        out[i] = (_Float16)in[i];
}

// ---------------- Main fused kernel ----------------
// One block = 64 rows (4 M-tiles of 16). 8 waves; wave w owns hidden columns
// [w*32, w*32+32) (2 column tiles x 4 gates). h0/x tiles staged in LDS so one
// B fragment feeds 4 WMMAs (one per M-tile) -> 4x less W_hh L2 traffic.
// LSTM elementwise is wave-local in f32 on TRANS units; h1 (f16) goes to LDS;
// then all 8 waves do the 16x16 projection tiles (4 M-tiles x 2 N-tiles).
__global__ __launch_bounds__(256) void traj_lstm_fused(
    const _Float16* __restrict__ xf,     // [N][32]   f16
    const _Float16* __restrict__ h0f,    // [N][256]  f16
    const float*    __restrict__ c0,     // [N][256]  f32
    const _Float16* __restrict__ wih,    // [1024][32]  f16  (W_ih row-major)
    const _Float16* __restrict__ whh,    // [1024][256] f16  (W_hh row-major)
    const _Float16* __restrict__ wpool,  // [32][256]   f16  (W_pool row-major)
    const float* __restrict__ b_ih, const float* __restrict__ b_hh,
    const float* __restrict__ b_pool,
    float* __restrict__ out)             // [N][32] f32
{
    __shared__ _Float16 h0s[MROWS][HID];   // 32 KB
    __shared__ _Float16 xs [MROWS][DOUT];  //  4 KB
    __shared__ _Float16 h1s[MROWS][HID];   // 32 KB

    const int tid  = threadIdx.x;
    const int wave = tid >> 5;
    const int lane = tid & 31;
    const int half = lane >> 4;
    const int l16  = lane & 15;
    const long row0 = (long)blockIdx.x * MROWS;

    // ---- cooperative tile fill ----
    // x tile: 64*32 f16 = 4 KB = 256 x 16B chunks (one per thread)
    {
        const uint4* src = (const uint4*)(xf + row0 * DOUT);
        uint4* dst = (uint4*)&xs[0][0];
        dst[tid] = src[tid];
    }
    // h0 tile: 64*256 f16 = 32 KB = 2048 x 16B chunks, async direct-to-LDS
#if USE_ASYNC_LDS
    {
        const unsigned lds_base = (unsigned)(size_t)&h0s[0][0];
        const unsigned long long gbase =
            (unsigned long long)(const void*)(h0f + row0 * HID);
#pragma unroll
        for (int it = 0; it < 8; ++it) {
            const int idx = it * 256 + tid;            // 16B chunk index
            unsigned lds_b = lds_base + idx * 16;
            unsigned long long ga = gbase + (unsigned long long)idx * 16ull;
            asm volatile("global_load_async_to_lds_b128 %0, %1, off"
                         :: "v"(lds_b), "v"(ga) : "memory");
        }
        asm volatile("s_wait_asynccnt 0" ::: "memory");
    }
#else
    {
        const uint4* src = (const uint4*)(h0f + row0 * HID);
        uint4* dst = (uint4*)&h0s[0][0];
#pragma unroll
        for (int it = 0; it < 8; ++it) dst[it * 256 + tid] = src[it * 256 + tid];
    }
#endif
    __syncthreads();

    // ---- gates GEMM + LSTM elementwise ----
#pragma unroll
    for (int t = 0; t < 2; ++t) {
        const int col = wave * 32 + t * 16 + l16;   // this lane's hidden column

        v8f acc[4][4];                               // [gate][mtile]
#pragma unroll
        for (int g = 0; g < 4; ++g)
#pragma unroll
            for (int m = 0; m < 4; ++m) acc[g][m] = v8f{};

        // x @ W_ih^T contribution (K=32)
        {
            v16h a[4];
#pragma unroll
            for (int m = 0; m < 4; ++m)
                a[m] = load_a(&xs[m * 16 + l16][0], half, 0);
#pragma unroll
            for (int g = 0; g < 4; ++g) {
                v16h b = load_b(wih, DOUT, g * HID + col, half, 0);
#pragma unroll
                for (int m = 0; m < 4; ++m)
                    acc[g][m] = __builtin_amdgcn_wmma_f32_16x16x32_f16(
                        false, a[m], false, b, (short)0, acc[g][m], false, false);
            }
        }
        // h0 @ W_hh^T contribution (K=256, 8 k-steps)
#pragma unroll
        for (int ko = 0; ko < 8; ++ko) {
            v16h a[4];
#pragma unroll
            for (int m = 0; m < 4; ++m)
                a[m] = load_a(&h0s[m * 16 + l16][0], half, ko * 32);
#pragma unroll
            for (int g = 0; g < 4; ++g) {
                v16h b = load_b(whh, HID, g * HID + col, half, ko * 32);
#pragma unroll
                for (int m = 0; m < 4; ++m)
                    acc[g][m] = __builtin_amdgcn_wmma_f32_16x16x32_f16(
                        false, a[m], false, b, (short)0, acc[g][m], false, false);
            }
        }

        // Gate biases depend only on the column -> one scalar per gate per lane
        float bias[4];
#pragma unroll
        for (int g = 0; g < 4; ++g)
            bias[g] = b_ih[g * HID + col] + b_hh[g * HID + col];

        // LSTM elementwise in f32 over C-layout fragments:
        // VGPR r holds row m*16 + r + 8*half, column col.
#pragma unroll
        for (int m = 0; m < 4; ++m) {
#pragma unroll
            for (int r = 0; r < 8; ++r) {
                const int  mrow = m * 16 + r + half * 8;
                const long rowg = row0 + mrow;
                float ig = acc[0][m][r] + bias[0];
                float fg = acc[1][m][r] + bias[1];
                float gg = acc[2][m][r] + bias[2];
                float og = acc[3][m][r] + bias[3];
                float c0v = c0[rowg * HID + col];
                float c1  = fast_sigmoid(fg) * c0v + fast_sigmoid(ig) * fast_tanh(gg);
                float h1  = fast_sigmoid(og) * fast_tanh(c1);
                h1s[mrow][col] = (_Float16)h1;
            }
        }
    }
    __syncthreads();

    // ---- output projection: out = h1 @ W_pool^T + b_pool ----
    // 64x32 per block = 8 16x16 tiles; wave w -> (mtile = w>>1, ntile = w&1).
    {
        const int mt = wave >> 1;
        const int nt = wave & 1;
        const int n  = nt * 16 + l16;
        v8f acc = v8f{};
#pragma unroll
        for (int ko = 0; ko < 8; ++ko) {
            v16h a = load_a(&h1s[mt * 16 + l16][0], half, ko * 32);
            v16h b = load_b(wpool, HID, n, half, ko * 32);
            acc = __builtin_amdgcn_wmma_f32_16x16x32_f16(
                false, a, false, b, (short)0, acc, false, false);
        }
        const float bp = b_pool[n];
#pragma unroll
        for (int r = 0; r < 8; ++r) {
            const int m = mt * 16 + r + half * 8;
            out[(row0 + m) * DOUT + n] = acc[r] + bp;
        }
    }
}

extern "C" void kernel_launch(void* const* d_in, const int* in_sizes, int n_in,
                              void* d_out, int out_size, void* d_ws, size_t ws_size,
                              hipStream_t stream) {
    const float* obs1   = (const float*)d_in[0];
    const float* obs2   = (const float*)d_in[1];
    const float* h0     = (const float*)d_in[2];
    const float* c0     = (const float*)d_in[3];
    const float* W_emb  = (const float*)d_in[4];
    const float* b_emb  = (const float*)d_in[5];
    const float* W_ih   = (const float*)d_in[6];
    const float* b_ih   = (const float*)d_in[7];
    const float* W_hh   = (const float*)d_in[8];
    const float* b_hh   = (const float*)d_in[9];
    const float* W_pool = (const float*)d_in[10];
    const float* b_pool = (const float*)d_in[11];
    float* out = (float*)d_out;

    char* ws = (char*)d_ws;
    // Workspace layout (bytes):
    float*    totals   = (float*)(ws + 0);          // 4 f32
    float*    partials = (float*)(ws + 256);        // 256*4 f32 = 4 KB
    _Float16* xf16     = (_Float16*)(ws + 8192);                 // N*32   f16 =  8 MB
    _Float16* h0f16    = (_Float16*)(ws + 8192 + 8388608);       // N*256  f16 = 64 MB
    _Float16* wih16    = (_Float16*)(ws + 8192 + 8388608 + 67108864);            // 64 KB
    _Float16* whh16    = (_Float16*)(ws + 8192 + 8388608 + 67108864 + 65536);    // 512 KB
    _Float16* wpool16  = (_Float16*)(ws + 8192 + 8388608 + 67108864 + 65536 + 524288); // 16 KB

    const int n = N_PED;

    // 1-2: deterministic global reduction for `total`
    traj_reduce1<<<256, 256, 0, stream>>>(obs1, obs2, partials, n);
    traj_reduce2<<<1, 32, 0, stream>>>(partials, totals);

    // 3: embedding MLP -> x (f16)
    traj_embed<<<(n + 255) / 256, 256, 0, stream>>>(obs1, obs2, totals,
                                                    W_emb, b_emb, xf16, n);

    // 4-7: f32 -> f16 staging (h0 + weights)
    traj_cvt_f16<<<2048, 256, 0, stream>>>(h0,     h0f16,   (long)n * HID);
    traj_cvt_f16<<<128,  256, 0, stream>>>(W_ih,   wih16,   (long)GATES * DOUT);
    traj_cvt_f16<<<1024, 256, 0, stream>>>(W_hh,   whh16,   (long)GATES * HID);
    traj_cvt_f16<<<32,   256, 0, stream>>>(W_pool, wpool16, (long)DOUT * HID);

    // 8: fused WMMA LSTM + pooling projection (64 rows per block)
    traj_lstm_fused<<<n / MROWS, 256, 0, stream>>>(xf16, h0f16, c0,
                                                   wih16, whh16, wpool16,
                                                   b_ih, b_hh, b_pool, out);
}
